// CIN_26680336842927
// MI455X (gfx1250) — matrix operand order repaired and checked by reading, not compiled, not run on hardware
//
#include <hip/hip_runtime.h>

// ---------------------------------------------------------------------------
// CIN (xDeepFM) on gfx1250 via f16 WMMA, software-pipelined.
//   x1[b,h,k] = sum_ij W1[h,i,j] x[b,i,k] x[b,j,k] + b1[h]
//   x2[b,h,k] = sum_ij W2[h,i,j] x1[b,i,k] x[b,j,k] + b2[h]
//   out[b, 0:128] = sum_k x1 ;  out[b, 128:256] = sum_k x2
// Per-batch GEMMs: A = flattened weights (pre-swizzled f16 in ws),
//                  B = outer-product matrix G staged in LDS (f16, double-
//                  buffered so formation overlaps WMMA across waves),
//                  C accumulated in f32 by v_wmma_f32_16x16x32_f16.
// ---------------------------------------------------------------------------

typedef _Float16 h2   __attribute__((ext_vector_type(2)));
typedef _Float16 v16h __attribute__((ext_vector_type(16)));
typedef float    v8f  __attribute__((ext_vector_type(8)));

#define MF      26          // fields
#define EK      32          // embedding size (N dimension of GEMMs)
#define H1N     128
#define NBATCH  2048

#define KT1     32          // layer-1 K-tiles: 32*32 = 1024 rows (676 real, rest 0)
#define KT2     104         // layer-2 K-tiles: 104*32 = 3328 = 128*26 rows exact
#define CHUNK   8           // K-tiles of G staged per pipeline step
#define NC1     (KT1 / CHUNK)   // 4
#define NC2     (KT2 / CHUNK)   // 13
#define TILE_HALFS  512     // one 16x32 (A) or 32x16 (B) f16 tile
#define TILE_DWORDS 256

#define W1_TILES (8 * KT1)  // 8 M-tiles x 32 K-tiles = 256
#define W2_TILES (8 * KT2)  // 8 M-tiles x 104 K-tiles = 832

// ---------------------------------------------------------------------------
// Weight conversion: f32 -> f16, pre-swizzled into the WMMA A-operand VGPR
// layout (16-bit A 16x32, ISA 7.12.2): lane l (row = l&15, kh = l>>4),
// vgpr v -> K = (v<4 ? 2v+8*kh : 16+2*(v-4)+8*kh) (+0/+1 for low/high half).
// One thread produces one packed dword (two consecutive K columns, same row).
// ---------------------------------------------------------------------------
__global__ void cin_convert_weights(const float* __restrict__ W1,
                                    const float* __restrict__ W2,
                                    _Float16* __restrict__ ws) {
    unsigned g = blockIdx.x * blockDim.x + threadIdx.x;
    const unsigned total = (unsigned)(W1_TILES + W2_TILES) * TILE_DWORDS;
    if (g >= total) return;

    unsigned t  = g >> 8;           // tile index
    unsigned s  = g & 255u;         // dword within tile
    unsigned ln = s >> 3;           // lane 0..31
    unsigned v  = s & 7u;           // vgpr 0..7
    unsigned kh = ln >> 4;
    unsigned r  = ln & 15u;
    unsigned k0 = (v < 4u) ? (2u*v + 8u*kh) : (16u + 2u*(v - 4u) + 8u*kh);

    float f0 = 0.0f, f1 = 0.0f;
    if (t < (unsigned)W1_TILES) {
        unsigned mt = t / KT1, kt = t % KT1;
        unsigned h  = mt * 16u + r;
        unsigned c0 = kt * 32u + k0;             // c = i*26 + j, real range [0,676)
        if (c0 < 676u)      f0 = W1[h * 676u + c0];
        if (c0 + 1u < 676u) f1 = W1[h * 676u + c0 + 1u];
    } else {
        unsigned t2 = t - (unsigned)W1_TILES;
        unsigned mt = t2 / KT2, kt = t2 % KT2;
        unsigned h  = mt * 16u + r;
        unsigned c0 = kt * 32u + k0;             // c = i*26 + j in [0,3328) always
        f0 = W2[h * 3328u + c0];
        f1 = W2[h * 3328u + c0 + 1u];
    }
    h2 p;
    p.x = (_Float16)f0;
    p.y = (_Float16)f1;
    ((h2*)ws)[g] = p;
}

// ---------------------------------------------------------------------------
// Build CHUNK k-tiles of G in LDS, already in the WMMA B-operand layout
// (32x16 16-bit B: lane = column, kh = lane>>4; vgpr v holds the pair
// K = kbase, kbase+1 with kbase = v<4 ? 2v+8*kh : 16+2*(v-4)+8*kh).
// G[(i*26+j), k] = A1[i,k] * A0[j,k], products computed in f32, stored f16.
// ---------------------------------------------------------------------------
template <bool LAYER1>
__device__ __forceinline__ void form_chunk(unsigned tid, unsigned kt0,
                                           const float* __restrict__ A1,
                                           const float* __restrict__ A0,
                                           unsigned* __restrict__ Gbuf) {
    for (unsigned d = tid; d < CHUNK * 2u * TILE_DWORDS; d += 256u) {
        unsigned t   = d >> 8;
        unsigned s   = d & 255u;
        unsigned ln  = s >> 3;
        unsigned v   = s & 7u;
        unsigned ktl = t >> 1;
        unsigned nt  = t & 1u;
        unsigned kt  = kt0 + ktl;
        unsigned row0 = kt * 32u + ((ln >> 4) << 3) + ((v & 3u) << 1) + ((v >> 2) << 4);
        unsigned col  = (nt << 4) + (ln & 15u);

        float p0 = 0.0f, p1 = 0.0f;
        if (!LAYER1 || row0 < 676u) {
            unsigned i = row0 / 26u, j = row0 - i * 26u;
            p0 = A1[i * 32u + col] * A0[j * 32u + col];
        }
        unsigned row1 = row0 + 1u;
        if (!LAYER1 || row1 < 676u) {
            unsigned i = row1 / 26u, j = row1 - i * 26u;
            p1 = A1[i * 32u + col] * A0[j * 32u + col];
        }
        h2 p;
        p.x = (_Float16)p0;
        p.y = (_Float16)p1;
        ((h2*)Gbuf)[d] = p;
    }
}

// Accumulate one chunk of WMMAs for this wave's M-tile (both N-tiles).
// Unroll capped at 4 to keep total VGPR usage below the 256 boundary
// (avoids s_set_vgpr_msb toggles between WMMAs, keeps occupancy up).
__device__ __forceinline__ void wmma_chunk(unsigned lane,
                                           const _Float16* __restrict__ wtiles,
                                           unsigned kt0,
                                           const unsigned* __restrict__ Gbuf,
                                           v8f& acc0, v8f& acc1) {
    const _Float16* GH = (const _Float16*)Gbuf;
    #pragma unroll 4
    for (unsigned ktl = 0; ktl < CHUNK; ++ktl) {
        v16h a  = *(const v16h*)(wtiles + (size_t)(kt0 + ktl) * TILE_HALFS + lane * 16u);
        v16h bA = *(const v16h*)(GH + (size_t)(ktl * 2u + 0u) * TILE_HALFS + lane * 16u);
        v16h bB = *(const v16h*)(GH + (size_t)(ktl * 2u + 1u) * TILE_HALFS + lane * 16u);
        acc0 = __builtin_amdgcn_wmma_f32_16x16x32_f16(false, a, false, bA,
                                                      (short)0, acc0, false, false);
        acc1 = __builtin_amdgcn_wmma_f32_16x16x32_f16(false, a, false, bB,
                                                      (short)0, acc1, false, false);
    }
}

// ---------------------------------------------------------------------------
// Main kernel: one workgroup (8 wave32) per batch element.
// Pipeline per layer (double-buffered G staging, one barrier per step):
//   form(0 -> buf0); barrier;
//   for c: wmma(c, buf[c&1]); if (c+1<NC) form(c+1 -> buf[(c+1)&1]); barrier;
// wmma(c) reads data formed before the previous barrier; form(c+1) overwrites
// a buffer whose last reader finished before the previous barrier.
// ---------------------------------------------------------------------------
__global__ __launch_bounds__(256)
void cin_main(const float* __restrict__ Xg,
              const float* __restrict__ b1g,
              const float* __restrict__ b2g,
              const _Float16* __restrict__ ws,
              float* __restrict__ out) {
    __shared__ float XLDS[MF * EK];                         //  3.3 KB
    __shared__ float X1LDS[H1N * EK];                       // 16.0 KB
    __shared__ float BIAS[256];                             //  1.0 KB
    __shared__ __align__(32) unsigned Gbuf[2][CHUNK * 2 * TILE_DWORDS]; // 2 x 16 KB

    const unsigned tid  = threadIdx.x;
    const unsigned wave = tid >> 5;     // 0..7 == output M-tile
    const unsigned lane = tid & 31u;
    const unsigned b    = blockIdx.x;

    for (unsigned i = tid; i < MF * EK; i += 256u)
        XLDS[i] = Xg[(size_t)b * (MF * EK) + i];
    BIAS[tid] = (tid < 128u) ? b1g[tid] : b2g[tid - 128u];
    __syncthreads();

    const unsigned kh = lane >> 4;
    const unsigned cl = lane & 15u;

    // ---------------- Layer 1: x1 = W1(128x1024) @ G1(1024x32) --------------
    {
        v8f acc0 = {}, acc1 = {};
        const _Float16* w1t = ws + (size_t)wave * KT1 * TILE_HALFS;
        form_chunk<true>(tid, 0u, XLDS, XLDS, Gbuf[0]);
        __syncthreads();
        for (unsigned c = 0; c < NC1; ++c) {
            wmma_chunk(lane, w1t, c * CHUNK, Gbuf[c & 1u], acc0, acc1);
            if (c + 1u < NC1)
                form_chunk<true>(tid, (c + 1u) * CHUNK, XLDS, XLDS, Gbuf[(c + 1u) & 1u]);
            __syncthreads();
        }
        // C/D layout: vgpr j, lane l -> row = j + 8*(l>>4), col = l&15
        #pragma unroll
        for (unsigned j = 0; j < 8; ++j) {
            unsigned h = wave * 16u + j + 8u * kh;
            float bias = BIAS[h];
            X1LDS[h * 32u + cl]       = acc0[j] + bias;
            X1LDS[h * 32u + 16u + cl] = acc1[j] + bias;
        }
    }
    __syncthreads();

    // ---------------- Layer 2: x2 = W2(128x3328) @ G2(3328x32) --------------
    float* X2 = (float*)Gbuf[0];    // reused after final barrier (4096 floats)
    {
        v8f acc0 = {}, acc1 = {};
        const _Float16* w2t = ws + (size_t)(W1_TILES + wave * KT2) * TILE_HALFS;
        form_chunk<false>(tid, 0u, X1LDS, XLDS, Gbuf[0]);
        __syncthreads();
        for (unsigned c = 0; c < NC2; ++c) {
            wmma_chunk(lane, w2t, c * CHUNK, Gbuf[c & 1u], acc0, acc1);
            if (c + 1u < NC2)
                form_chunk<false>(tid, (c + 1u) * CHUNK, X1LDS, XLDS, Gbuf[(c + 1u) & 1u]);
            __syncthreads();
        }
        #pragma unroll
        for (unsigned j = 0; j < 8; ++j) {
            unsigned h = wave * 16u + j + 8u * kh;
            float bias = BIAS[128u + h];
            X2[h * 32u + cl]       = acc0[j] + bias;
            X2[h * 32u + 16u + cl] = acc1[j] + bias;
        }
    }
    __syncthreads();

    // ---------------- Reduce over k, write [b, 0:256] ------------------------
    {
        const float* src = (tid < 128u) ? &X1LDS[tid * 32u] : &X2[(tid - 128u) * 32u];
        float s = 0.0f;
        #pragma unroll
        for (unsigned k = 0; k < EK; ++k) s += src[k];
        out[(size_t)b * 256u + tid] = s;
    }
}

// ---------------------------------------------------------------------------
extern "C" void kernel_launch(void* const* d_in, const int* in_sizes, int n_in,
                              void* d_out, int out_size, void* d_ws, size_t ws_size,
                              hipStream_t stream) {
    (void)in_sizes; (void)n_in; (void)out_size; (void)ws_size;
    const float* X  = (const float*)d_in[0];   // [2048, 26, 32]
    const float* W1 = (const float*)d_in[1];   // [128, 26, 26]
    const float* b1 = (const float*)d_in[2];   // [128]
    const float* W2 = (const float*)d_in[3];   // [128, 128, 26]
    const float* b2 = (const float*)d_in[4];   // [128]
    float* out      = (float*)d_out;           // [2048, 256]
    _Float16* ws    = (_Float16*)d_ws;         // needs (256+832)*1KB ~ 1.06 MB

    const unsigned totalDwords = (unsigned)(W1_TILES + W2_TILES) * TILE_DWORDS;
    cin_convert_weights<<<(totalDwords + 255u) / 256u, 256, 0, stream>>>(W1, W2, ws);
    cin_main<<<NBATCH, 256, 0, stream>>>(X, b1, b2, ws, out);
}
